// Transformer_14276471292540
// MI455X (gfx1250) — matrix operand order repaired and checked by reading, not compiled, not run on hardware
//
#include <hip/hip_runtime.h>

#define DEV __device__ __forceinline__

typedef _Float16 v8h  __attribute__((ext_vector_type(8)));
typedef _Float16 v16h __attribute__((ext_vector_type(16)));
typedef float    v8f  __attribute__((ext_vector_type(8)));
typedef unsigned int v4u __attribute__((ext_vector_type(4)));
typedef int v4i __attribute__((ext_vector_type(4)));
typedef int v8i __attribute__((ext_vector_type(8)));

#if __has_builtin(__builtin_amdgcn_tensor_load_to_lds)
#define HAVE_TDM 1
#else
#define HAVE_TDM 0
#endif

DEV v16h cat16(v8h lo, v8h hi) {
  return __builtin_shufflevector(lo, hi, 0,1,2,3,4,5,6,7,8,9,10,11,12,13,14,15);
}

// A-fragment: 16xK row-major tile [m][k] in LDS (stride in halves).
// Lane m=lane&15; K = 8*kh..+7 (VGPR0-3) and 16+8*kh..+7 (VGPR4-7), kh=lane>>4.
DEV v16h load_fragA(const _Float16* base, int stride) {
  int lane = threadIdx.x & 31;
  const _Float16* p = base + (lane & 15) * stride + (lane >> 4) * 8;
  v8h lo = *(const v8h*)(p);
  v8h hi = *(const v8h*)(p + 16);
  return cat16(lo, hi);
}

// B-fragment: Kx16 tile stored transposed [n][k] in LDS (stride in halves).
// Lane n=lane&15; K = 8*kh..+7 (VGPR0-3) and 16+8*kh..+7 (VGPR4-7), kh=lane>>4.
DEV v16h load_fragB(const _Float16* base, int stride) {
  int lane = threadIdx.x & 31;
  const _Float16* p = base + (lane & 15) * stride + (lane >> 4) * 8;
  v8h lo = *(const v8h*)(p);
  v8h hi = *(const v8h*)(p + 16);
  return cat16(lo, hi);
}

DEV v8f wmma_f16(v16h a, v16h b, v8f c) {
  return __builtin_amdgcn_wmma_f32_16x16x32_f16(false, a, false, b, (short)0, c,
                                                false, false);
}

#if HAVE_TDM
// Issue one TDM 2D f16 tile load: tileD0 contiguous elems per row, tileD1 rows,
// row stride strideElems (all in elements).  Tensor dims == tile dims -> no OOB.
// D# layout per CDNA5 ISA ch.8 (group0 128b, group1 256b; groups 2/3 zero = 2D).
// This toolchain's builtin takes 6 args: (g0 v4u, g1 v8i, g2 v4i, g3 v4i, v8i, cpol).
DEV void tdm_load_f16_2d(_Float16* lds, const _Float16* g,
                         int tileD0, int tileD1, int strideElems) {
  unsigned lds_off = (unsigned)(size_t)lds;            // low 32b of flat = LDS addr
  unsigned long long ga = (unsigned long long)(size_t)g;
  v4u g0;
  g0.x = 1u;                                           // count=1 (valid user D#)
  g0.y = lds_off;                                      // lds_addr [63:32]
  g0.z = (unsigned)(ga & 0xffffffffu);                 // global_addr[31:0]
  g0.w = (unsigned)((ga >> 32) & 0x01ffffffu) | (2u << 30);  // addr[56:32] | type=2
  v8i g1;
  g1[0] = 1 << 16;                                     // data_size=1 (2 bytes)
  g1[1] = (tileD0 & 0xffff) << 16;                     // tensor_dim0[15:0] @63:48
  g1[2] = ((unsigned)tileD0 >> 16) | ((tileD1 & 0xffff) << 16);  // dim0 hi | dim1 lo
  g1[3] = ((unsigned)tileD1 >> 16) | ((tileD0 & 0xffff) << 16);  // dim1 hi | tile_dim0
  g1[4] = tileD1 & 0xffff;                             // tile_dim1 (tile_dim2=0)
  g1[5] = strideElems;                                 // tensor_dim0_stride[31:0]
  g1[6] = 0;
  g1[7] = 0;
  v4i z4 = {0, 0, 0, 0};
  v8i z8 = {0, 0, 0, 0, 0, 0, 0, 0};
  __builtin_amdgcn_tensor_load_to_lds(g0, g1, z4, z4, z8, 0);
}
#endif

// ---------------------------------------------------------------------------
// Weight pre-pass: Wt[n*K + k] = (f16) W[k*N + n]
// ---------------------------------------------------------------------------
__global__ void wt_kernel(_Float16* __restrict__ o, const float* __restrict__ w,
                          int K, int N, int total) {
  int j = blockIdx.x * 256 + threadIdx.x;
  if (j < total) {
    int n = j / K, k = j - n * K;
    o[j] = (_Float16)w[(size_t)k * N + n];
  }
}

// ---------------------------------------------------------------------------
// Attention pre-pass: pack head-major f16 tensors with zero padding.
// ---------------------------------------------------------------------------
__global__ void pack_head(_Float16* __restrict__ o, const float* __restrict__ x,
                          int S, int Sp, int total) {
  int i = blockIdx.x * 256 + threadIdx.x;
  if (i >= total) return;
  int c = i & 63;
  int sp = (i >> 6) % Sp;
  int bh = i / (Sp * 64);
  int b = bh >> 3, h = bh & 7;
  float v = (sp < S) ? x[(size_t)(b * S + sp) * 512 + h * 64 + c] : 0.f;
  o[i] = (_Float16)v;
}
__global__ void pack_head_t(_Float16* __restrict__ o, const float* __restrict__ x,
                            int S, int Sp, int total) {
  int i = blockIdx.x * 256 + threadIdx.x;
  if (i >= total) return;
  int sp = i % Sp;
  int c = (i / Sp) & 63;
  int bh = i / (Sp * 64);
  int b = bh >> 3, h = bh & 7;
  float v = (sp < S) ? x[(size_t)(b * S + sp) * 512 + h * 64 + c] : 0.f;
  o[i] = (_Float16)v;
}

// ---------------------------------------------------------------------------
// GEMM: C[M,N] = act(A[M,K] @ W[K,N] + bias[N]); W pre-converted to f16 Wt[N][K].
// Block tile 128x128, 8 waves, wave = 16x128 strip.  Double-buffered LDS:
// A prefetched through VGPRs, B streamed by the Tensor Data Mover (wave 0
// issues, drains TENSORcnt before the barrier).  One barrier per K-step.
// ---------------------------------------------------------------------------
__global__ __launch_bounds__(256) void gemm_kernel(
    const float* __restrict__ A, const _Float16* __restrict__ Wt,
    const float* __restrict__ bias, float* __restrict__ C,
    int M, int N, int K, int relu)
{
  __shared__ __align__(16) _Float16 sA[2][128 * 32];  // [m][k]
  __shared__ __align__(16) _Float16 sB[2][128 * 32];  // [n][k]
  const int tid = threadIdx.x, lane = tid & 31, wave = tid >> 5;
  const int m0 = blockIdx.y * 128, n0 = blockIdx.x * 128;
  const int nsteps = K >> 5;

  const int ar = tid >> 1, ac = (tid & 1) * 16;   // this thread's A-fill slot
  const int agm = m0 + ar;
  const bool avalid = agm < M;

  v8f zf = {0.f,0.f,0.f,0.f,0.f,0.f,0.f,0.f};
  v8f acc[8];
  #pragma unroll
  for (int j = 0; j < 8; ++j) acc[j] = zf;

  // ---- prologue: tile 0 ----
  {
    _Float16* dst = sA[0] + ar * 32 + ac;
    if (avalid) {
      const float* s = A + (size_t)agm * K + ac;
      #pragma unroll
      for (int i = 0; i < 16; ++i) dst[i] = (_Float16)s[i];
    } else {
      #pragma unroll
      for (int i = 0; i < 16; ++i) dst[i] = (_Float16)0.f;
    }
  }
#if HAVE_TDM
  if (wave == 0) {
    tdm_load_f16_2d(sB[0], Wt + (size_t)n0 * K, 32, 128, K);
    __builtin_amdgcn_s_wait_tensorcnt(0);
  }
#else
  {
    const _Float16* src = Wt + (size_t)(n0 + ar) * K + ac;
    v8h* d = (v8h*)(sB[0] + ar * 32 + ac);
    d[0] = *(const v8h*)(src);
    d[1] = *(const v8h*)(src + 8);
  }
#endif
  __syncthreads();

  for (int i = 0; i < nsteps; ++i) {
    const int cur = i & 1, nxt = cur ^ 1;
    const int k0n = (i + 1) << 5;
    const bool more = (i + 1) < nsteps;

    float areg[16];
    if (more) {
      if (avalid) {  // prefetch next A tile into VGPRs (overlaps wmma below)
        const float* s = A + (size_t)agm * K + k0n + ac;
        #pragma unroll
        for (int j = 0; j < 16; ++j) areg[j] = s[j];
      }
#if HAVE_TDM
      if (wave == 0)  // async DMA next B tile into the other buffer
        tdm_load_f16_2d(sB[nxt], Wt + (size_t)n0 * K + k0n, 32, 128, K);
#endif
    }

    v16h a = load_fragA(sA[cur] + (wave * 16) * 32, 32);
    #pragma unroll
    for (int j = 0; j < 8; ++j) {
      v16h b = load_fragB(sB[cur] + (j * 16) * 32, 32);
      acc[j] = wmma_f16(a, b, acc[j]);
    }

    if (more) {
      _Float16* dst = sA[nxt] + ar * 32 + ac;
      if (avalid) {
        #pragma unroll
        for (int j = 0; j < 16; ++j) dst[j] = (_Float16)areg[j];
      } else {
        #pragma unroll
        for (int j = 0; j < 16; ++j) dst[j] = (_Float16)0.f;
      }
#if !HAVE_TDM
      const _Float16* src = Wt + (size_t)(n0 + ar) * K + k0n + ac;
      v8h* d = (v8h*)(sB[nxt] + ar * 32 + ac);
      d[0] = *(const v8h*)(src);
      d[1] = *(const v8h*)(src + 8);
#else
      if (wave == 0) __builtin_amdgcn_s_wait_tensorcnt(0);
#endif
    }
    __syncthreads();
  }

  const int n = lane & 15, half = lane >> 4;
  #pragma unroll
  for (int j = 0; j < 8; ++j)
    #pragma unroll
    for (int r = 0; r < 8; ++r) {
      int row = m0 + wave * 16 + half * 8 + r;
      int col = n0 + j * 16 + n;
      if (row < M) {
        float v = acc[j][r] + bias[col];
        if (relu) v = fmaxf(v, 0.f);
        C[(size_t)row * N + col] = v;
      }
    }
}

// ---------------------------------------------------------------------------
// Flash attention on pre-packed f16 heads. dk=64.
//   Qh/Kh: [B*H, Sp, 64], Vt: [B*H, 64, Sp]  (pad rows zeroed)
// Grid: (ceil(Sq/128), H, B); block 256 = 8 waves, wave owns 16 query rows.
// ---------------------------------------------------------------------------
__global__ __launch_bounds__(256) void attn_kernel(
    const _Float16* __restrict__ Qh, const _Float16* __restrict__ Kh,
    const _Float16* __restrict__ Vt, float* __restrict__ O,
    int Sq, int Sk, int Sqp, int Skp)
{
  const int b = blockIdx.z, h = blockIdx.y, bh = b * 8 + h;
  const int tid = threadIdx.x, lane = tid & 31, wave = tid >> 5;
  const int q0 = blockIdx.x * 128 + wave * 16;

  __shared__ __align__(16) _Float16 sQ[8][16 * 64];  // per-wave [m][dk]
  __shared__ __align__(16) _Float16 sK[32 * 64];     // [key][dk]
  __shared__ __align__(16) _Float16 sV[64 * 32];     // [dk][key]
  __shared__ __align__(16) float    sS[8][16 * 32];  // raw scores
  __shared__ __align__(16) _Float16 sP[8][16 * 32];  // probs
  __shared__ float sM[8][16], sL[8][16], sC[8][16];

  { // Q tile copy (zero-padded at pack time)
    int r = lane >> 1, c0 = (lane & 1) * 32;
    const v8h* s = (const v8h*)(Qh + ((size_t)bh * Sqp + q0 + r) * 64 + c0);
    v8h* d = (v8h*)(sQ[wave] + r * 64 + c0);
    d[0] = s[0]; d[1] = s[1]; d[2] = s[2]; d[3] = s[3];
  }
  if (lane < 16) { sM[wave][lane] = -1e30f; sL[wave][lane] = 0.f; }

  v8f zf = {0.f,0.f,0.f,0.f,0.f,0.f,0.f,0.f};
  v8f accO[4];
  #pragma unroll
  for (int j = 0; j < 4; ++j) accO[j] = zf;

  for (int kt = 0; kt < Sk; kt += 32) {
    __syncthreads();  // protect sK/sV (and first-iter sQ/stats)
    { // K tile copy
      int key = tid >> 3, c = (tid & 7) * 8;
      *(v8h*)(sK + key * 64 + c) =
          *(const v8h*)(Kh + ((size_t)bh * Skp + kt + key) * 64 + c);
    }
    { // V^T tile copy
      int row = tid >> 2, seg = (tid & 3) * 8;
      *(v8h*)(sV + row * 32 + seg) =
          *(const v8h*)(Vt + ((size_t)bh * 64 + row) * Skp + kt + seg);
    }
    __syncthreads();

    // scores: S[16x32] = Q[16x64] @ K^T[64x32]
    v8f accS[2] = {zf, zf};
    #pragma unroll
    for (int kk = 0; kk < 64; kk += 32) {
      v16h aQ = load_fragA(sQ[wave] + kk, 64);
      accS[0] = wmma_f16(aQ, load_fragB(sK + kk, 64), accS[0]);
      accS[1] = wmma_f16(aQ, load_fragB(sK + 16 * 64 + kk, 64), accS[1]);
    }
    { // spill scaled+masked scores
      int n = lane & 15, half = lane >> 4;
      #pragma unroll
      for (int j = 0; j < 2; ++j)
        #pragma unroll
        for (int r = 0; r < 8; ++r) {
          int row = half * 8 + r, col = j * 16 + n;
          float s = accS[j][r] * 0.125f;
          if (kt + col >= Sk) s = -1e30f;
          sS[wave][row * 32 + col] = s;
        }
    }
    __syncthreads();

    if (lane < 16) { // online softmax, one lane per query row
      int row = lane;
      float* sr = sS[wave] + row * 32;
      float mold = sM[wave][row];
      float mnew = mold;
      #pragma unroll
      for (int j = 0; j < 32; ++j) mnew = fmaxf(mnew, sr[j]);
      _Float16* pr = sP[wave] + row * 32;
      float tsum = 0.f;
      #pragma unroll
      for (int j = 0; j < 32; ++j) {
        float p = __expf(sr[j] - mnew);
        tsum += p;
        pr[j] = (_Float16)p;
      }
      float sc = __expf(mold - mnew);
      sL[wave][row] = sL[wave][row] * sc + tsum;
      sM[wave][row] = mnew;
      sC[wave][row] = sc;
    }
    __syncthreads();

    { // rescale running output, accumulate P[16x32] @ V[32x64]
      int half = lane >> 4;
      #pragma unroll
      for (int j = 0; j < 4; ++j)
        #pragma unroll
        for (int r = 0; r < 8; ++r) accO[j][r] *= sC[wave][half * 8 + r];
      v16h aP = load_fragA(sP[wave], 32);
      #pragma unroll
      for (int j = 0; j < 4; ++j)
        accO[j] = wmma_f16(aP, load_fragB(sV + (j * 16) * 32, 32), accO[j]);
    }
  }

  { // normalize and store
    int n = lane & 15, half = lane >> 4;
    #pragma unroll
    for (int j = 0; j < 4; ++j)
      #pragma unroll
      for (int r = 0; r < 8; ++r) {
        int row = q0 + half * 8 + r;
        if (row < Sq) {
          float inv = 1.f / sL[wave][half * 8 + r];
          O[(size_t)(b * Sq + row) * 512 + h * 64 + j * 16 + n] = accO[j][r] * inv;
        }
      }
  }
}

// ---------------------------------------------------------------------------
// LayerNorm with fused residual: Out[row] = LN(A[row] + R[row]) * g + b.
// ---------------------------------------------------------------------------
__global__ __launch_bounds__(256) void ln_kernel(
    const float* __restrict__ A, const float* __restrict__ R,
    const float* __restrict__ g, const float* __restrict__ be,
    float* __restrict__ Out, int M)
{
  const int wave = threadIdx.x >> 5, lane = threadIdx.x & 31;
  const int row = blockIdx.x * 8 + wave;
  if (row >= M) return;
  const float* a = A + (size_t)row * 512;
  const float* r = R + (size_t)row * 512;
  float x[16], s = 0.f;
  #pragma unroll
  for (int i = 0; i < 16; ++i) {
    x[i] = a[lane + i * 32] + r[lane + i * 32];
    s += x[i];
  }
  #pragma unroll
  for (int off = 16; off; off >>= 1) s += __shfl_xor(s, off, 32);
  float mean = s * (1.f / 512.f), v = 0.f;
  #pragma unroll
  for (int i = 0; i < 16; ++i) { float d = x[i] - mean; v += d * d; }
  #pragma unroll
  for (int off = 16; off; off >>= 1) v += __shfl_xor(v, off, 32);
  float rstd = rsqrtf(v * (1.f / 512.f) + 1e-5f);
  #pragma unroll
  for (int i = 0; i < 16; ++i) {
    int c = lane + i * 32;
    Out[(size_t)row * 512 + c] = (x[i] - mean) * rstd * g[c] + be[c];
  }
}

// ---------------------------------------------------------------------------
// Elementwise helpers
// ---------------------------------------------------------------------------
__global__ void add2_kernel(float* o, const float* a, const float* b, int n) {
  int i = blockIdx.x * 256 + threadIdx.x;
  if (i < n) o[i] = a[i] + b[i];
}
__global__ void addq_kernel(float* o, const float* a, const float* qe, int n) {
  int i = blockIdx.x * 256 + threadIdx.x;
  if (i < n) {
    int d = i & 511;
    int nq = (i >> 9) % 100;
    o[i] = a[i] + qe[nq * 512 + d];
  }
}
__global__ void transpose_kernel(float* o, const float* in, int C, int S, int n) {
  int i = blockIdx.x * 256 + threadIdx.x;
  if (i < n) {
    int c = i % C;
    int s = (i / C) % S;
    int b = i / (C * S);
    o[i] = in[((size_t)b * C + c) * S + s];
  }
}
__global__ void zero_kernel(float* o, int n) {
  int i = blockIdx.x * 256 + threadIdx.x;
  if (i < n) o[i] = 0.f;
}
__global__ void copy_kernel(float* o, const float* a, int n) {
  int i = blockIdx.x * 256 + threadIdx.x;
  if (i < n) o[i] = a[i];
}

// ---------------------------------------------------------------------------
// Orchestration
// ---------------------------------------------------------------------------
static inline dim3 g1(size_t n) { return dim3((unsigned)((n + 255) / 256)); }

extern "C" void kernel_launch(void* const* d_in, const int* in_sizes, int n_in,
                              void* d_out, int out_size, void* d_ws, size_t ws_size,
                              hipStream_t stream) {
  (void)in_sizes; (void)n_in; (void)out_size; (void)ws_size;
  const float* src  = (const float*)d_in[0];
  const float* qe   = (const float*)d_in[1];
  const float* pos  = (const float*)d_in[2];
  const float* eaw  = (const float*)d_in[3];
  const float* eab  = (const float*)d_in[4];
  const float* ew1  = (const float*)d_in[5];
  const float* eb1  = (const float*)d_in[6];
  const float* ew2  = (const float*)d_in[7];
  const float* eb2  = (const float*)d_in[8];
  const float* elnw = (const float*)d_in[9];
  const float* elnb = (const float*)d_in[10];
  const float* saw  = (const float*)d_in[11];
  const float* sab  = (const float*)d_in[12];
  const float* caw  = (const float*)d_in[13];
  const float* cab  = (const float*)d_in[14];
  const float* dw1  = (const float*)d_in[15];
  const float* db1  = (const float*)d_in[16];
  const float* dw2  = (const float*)d_in[17];
  const float* db2  = (const float*)d_in[18];
  const float* dlnw = (const float*)d_in[19];
  const float* dlnb = (const float*)d_in[20];

  const int Bb = 4, S = 1024, Dm = 512, FFd = 2048, NQ = 100, Ll = 6;
  const int Me = Bb * S;
  const int Md = Bb * NQ;
  const int NQp = 128;
  const size_t E  = (size_t)Me * Dm;
  const size_t Dq = (size_t)Md * Dm;
  const size_t Fq = (size_t)Md * FFd;
  const size_t DD = (size_t)Dm * Dm;
  const size_t DF = (size_t)Dm * FFd;

  float* ws    = (float*)d_ws;
  float* posT  = ws;
  float* mem   = ws + 1 * E;
  float* t     = ws + 2 * E;
  float* qb    = ws + 3 * E;
  float* kb    = ws + 4 * E;
  float* vb    = ws + 5 * E;
  float* ao    = ws + 6 * E;
  float* pr    = ws + 7 * E;
  float* memin = ws + 8 * E;
  float* ffn   = ws + 9 * E;    // [B,S,FF] = 4E
  float* outD  = ws + 13 * E;
  float* td    = outD + 1 * Dq;
  float* qd    = outD + 2 * Dq;
  float* aod   = outD + 3 * Dq;
  float* prd   = outD + 4 * Dq;
  float* ffnd  = outD + 5 * Dq;
  _Float16* wh = (_Float16*)(ffnd + Fq);
  const size_t encStride = 4 * DD + 2 * DF;
  const size_t decStride = 8 * DD + 2 * DF;
  _Float16* encW = wh;
  _Float16* decW = wh + Ll * encStride;
  _Float16* Qh = decW + Ll * decStride;
  _Float16* Kh = Qh + (size_t)32 * 1024 * 64;
  _Float16* Vh = Kh + (size_t)32 * 1024 * 64;

  dim3 blk(256);
  dim3 gG512e(Dm / 128, (Me + 127) / 128);
  dim3 gGffE(FFd / 128, (Me + 127) / 128);
  dim3 gG512d(Dm / 128, (Md + 127) / 128);
  dim3 gGffD(FFd / 128, (Md + 127) / 128);
  dim3 gAe((S + 127) / 128, 8, Bb);
  dim3 gAd((NQ + 127) / 128, 8, Bb);
  const int PKe = 32 * 1024 * 64;
  const int PKd = 32 * NQp * 64;

  // ---------------- weight f32->f16 transpose pre-pass ----------------
  for (int l = 0; l < Ll; ++l) {
    _Float16* base = encW + (size_t)l * encStride;
    for (int j = 0; j < 4; ++j)
      wt_kernel<<<g1(DD), blk, 0, stream>>>(base + j * DD, eaw + ((size_t)l * 4 + j) * DD,
                                            Dm, Dm, (int)DD);
    wt_kernel<<<g1(DF), blk, 0, stream>>>(base + 4 * DD,      ew1 + (size_t)l * DF, Dm, FFd, (int)DF);
    wt_kernel<<<g1(DF), blk, 0, stream>>>(base + 4 * DD + DF, ew2 + (size_t)l * DF, FFd, Dm, (int)DF);
  }
  for (int l = 0; l < Ll; ++l) {
    _Float16* base = decW + (size_t)l * decStride;
    for (int j = 0; j < 4; ++j)
      wt_kernel<<<g1(DD), blk, 0, stream>>>(base + j * DD, saw + ((size_t)l * 4 + j) * DD,
                                            Dm, Dm, (int)DD);
    for (int j = 0; j < 4; ++j)
      wt_kernel<<<g1(DD), blk, 0, stream>>>(base + (4 + j) * DD, caw + ((size_t)l * 4 + j) * DD,
                                            Dm, Dm, (int)DD);
    wt_kernel<<<g1(DF), blk, 0, stream>>>(base + 8 * DD,      dw1 + (size_t)l * DF, Dm, FFd, (int)DF);
    wt_kernel<<<g1(DF), blk, 0, stream>>>(base + 8 * DD + DF, dw2 + (size_t)l * DF, FFd, Dm, (int)DF);
  }

  transpose_kernel<<<g1(E), blk, 0, stream>>>(mem,  src, Dm, S, (int)E);
  transpose_kernel<<<g1(E), blk, 0, stream>>>(posT, pos, Dm, S, (int)E);

  // ---------------- encoder ----------------
  for (int l = 0; l < Ll; ++l) {
    _Float16* aw = encW + (size_t)l * encStride;
    const float* ab = eab + (size_t)l * 4 * Dm;
    add2_kernel<<<g1(E), blk, 0, stream>>>(t, mem, posT, (int)E);
    gemm_kernel<<<gG512e, blk, 0, stream>>>(t, aw,          ab,          qb, Me, Dm, Dm, 0);
    gemm_kernel<<<gG512e, blk, 0, stream>>>(t, aw + DD,     ab + Dm,     kb, Me, Dm, Dm, 0);
    gemm_kernel<<<gG512e, blk, 0, stream>>>(t, aw + 2 * DD, ab + 2 * Dm, vb, Me, Dm, Dm, 0);
    pack_head  <<<g1(PKe), blk, 0, stream>>>(Qh, qb, S, S, PKe);
    pack_head  <<<g1(PKe), blk, 0, stream>>>(Kh, kb, S, S, PKe);
    pack_head_t<<<g1(PKe), blk, 0, stream>>>(Vh, vb, S, S, PKe);
    attn_kernel<<<gAe, blk, 0, stream>>>(Qh, Kh, Vh, ao, S, S, S, S);
    gemm_kernel<<<gG512e, blk, 0, stream>>>(ao, aw + 3 * DD, ab + 3 * Dm, pr, Me, Dm, Dm, 0);
    ln_kernel<<<dim3((Me + 7) / 8), blk, 0, stream>>>(
        t, pr, elnw + (size_t)(l * 2) * Dm, elnb + (size_t)(l * 2) * Dm, t, Me);
    gemm_kernel<<<gGffE, blk, 0, stream>>>(t,   aw + 4 * DD,       eb1 + (size_t)l * FFd, ffn, Me, FFd, Dm, 1);
    gemm_kernel<<<gG512e, blk, 0, stream>>>(ffn, aw + 4 * DD + DF, eb2 + (size_t)l * Dm,  pr,  Me, Dm, FFd, 0);
    ln_kernel<<<dim3((Me + 7) / 8), blk, 0, stream>>>(
        t, pr, elnw + (size_t)(l * 2 + 1) * Dm, elnb + (size_t)(l * 2 + 1) * Dm, mem, Me);
  }

  add2_kernel<<<g1(E), blk, 0, stream>>>(memin, mem, posT, (int)E);
  zero_kernel<<<g1(Dq), blk, 0, stream>>>(outD, (int)Dq);

  // ---------------- decoder ----------------
  for (int l = 0; l < Ll; ++l) {
    _Float16* sw = decW + (size_t)l * decStride;
    _Float16* cw = sw + 4 * DD;
    const float* sb = sab + (size_t)l * 4 * Dm;
    const float* cb = cab + (size_t)l * 4 * Dm;
    addq_kernel<<<g1(Dq), blk, 0, stream>>>(td, outD, qe, (int)Dq);
    gemm_kernel<<<gG512d, blk, 0, stream>>>(td, sw,          sb,          qd, Md, Dm, Dm, 0);
    gemm_kernel<<<gG512d, blk, 0, stream>>>(td, sw + DD,     sb + Dm,     kb, Md, Dm, Dm, 0);
    gemm_kernel<<<gG512d, blk, 0, stream>>>(td, sw + 2 * DD, sb + 2 * Dm, vb, Md, Dm, Dm, 0);
    pack_head  <<<g1(PKd), blk, 0, stream>>>(Qh, qd, NQ, NQp, PKd);
    pack_head  <<<g1(PKd), blk, 0, stream>>>(Kh, kb, NQ, NQp, PKd);
    pack_head_t<<<g1(PKd), blk, 0, stream>>>(Vh, vb, NQ, NQp, PKd);
    attn_kernel<<<gAd, blk, 0, stream>>>(Qh, Kh, Vh, aod, NQ, NQ, NQp, NQp);
    gemm_kernel<<<gG512d, blk, 0, stream>>>(aod, sw + 3 * DD, sb + 3 * Dm, prd, Md, Dm, Dm, 0);
    ln_kernel<<<dim3((Md + 7) / 8), blk, 0, stream>>>(
        td, prd, dlnw + (size_t)(l * 3) * Dm, dlnb + (size_t)(l * 3) * Dm, td, Md);
    gemm_kernel<<<gG512d, blk, 0, stream>>>(td,    cw,          cb,          qd, Md, Dm, Dm, 0);
    gemm_kernel<<<gG512e, blk, 0, stream>>>(memin, cw + DD,     cb + Dm,     kb, Me, Dm, Dm, 0);
    gemm_kernel<<<gG512e, blk, 0, stream>>>(memin, cw + 2 * DD, cb + 2 * Dm, vb, Me, Dm, Dm, 0);
    pack_head  <<<g1(PKd), blk, 0, stream>>>(Qh, qd, NQ, NQp, PKd);
    pack_head  <<<g1(PKe), blk, 0, stream>>>(Kh, kb, S, S, PKe);
    pack_head_t<<<g1(PKe), blk, 0, stream>>>(Vh, vb, S, S, PKe);
    attn_kernel<<<gAd, blk, 0, stream>>>(Qh, Kh, Vh, aod, NQ, S, NQp, S);
    gemm_kernel<<<gG512d, blk, 0, stream>>>(aod, cw + 3 * DD, cb + 3 * Dm, prd, Md, Dm, Dm, 0);
    ln_kernel<<<dim3((Md + 7) / 8), blk, 0, stream>>>(
        td, prd, dlnw + (size_t)(l * 3 + 1) * Dm, dlnb + (size_t)(l * 3 + 1) * Dm, td, Md);
    gemm_kernel<<<gGffD, blk, 0, stream>>>(td,   sw + 8 * DD,       db1 + (size_t)l * FFd, ffnd, Md, FFd, Dm, 1);
    gemm_kernel<<<gG512d, blk, 0, stream>>>(ffnd, sw + 8 * DD + DF, db2 + (size_t)l * Dm,  prd,  Md, Dm, FFd, 0);
    ln_kernel<<<dim3((Md + 7) / 8), blk, 0, stream>>>(
        td, prd, dlnw + (size_t)(l * 3 + 2) * Dm, dlnb + (size_t)(l * 3 + 2) * Dm, outD, Md);
  }

  float* out = (float*)d_out;
  copy_kernel<<<g1(Dq), blk, 0, stream>>>(out, outD, (int)Dq);
  copy_kernel<<<g1(E), blk, 0, stream>>>(out + Dq, mem, (int)E);
}
